// ClassicalGNN_58574763983391
// MI455X (gfx1250) — compile-verified, matrix operand order checked by reading, not run on hardware
//
#include <hip/hip_runtime.h>
#include <hip/hip_bf16.h>

#define NN 100000
#define EE 1200000
#define GG 512
#define HH 64

typedef __attribute__((ext_vector_type(16))) _Float16 v16h;
typedef __attribute__((ext_vector_type(8)))  _Float16 v8h;
typedef __attribute__((ext_vector_type(8)))  float    v8f;

// ---------------------------------------------------------------------------
// 1. Node embedding lookup (5 tables, concat->100) + projection to H=64
// ---------------------------------------------------------------------------
__global__ __launch_bounds__(256) void k_embed_proj(
    const int* __restrict__ x,
    const float* __restrict__ e0, const float* __restrict__ e1,
    const float* __restrict__ e2, const float* __restrict__ e3,
    const float* __restrict__ e4,
    const float* __restrict__ pw, const float* __restrict__ pb,
    float* __restrict__ h)
{
    __shared__ float feat[4][100];
    const int ln   = threadIdx.x >> 6;
    const int c    = threadIdx.x & 63;
    const int node = blockIdx.x * 4 + ln;
    if (node < NN) {
        for (int k = c; k < 100; k += 64) {
            int ti, off, vocab, dd; const float* tab;
            if      (k < 64) { ti = 0; off = k;      tab = e0; vocab = 120; dd = 64; }
            else if (k < 80) { ti = 1; off = k - 64; tab = e1; vocab = 10;  dd = 16; }
            else if (k < 88) { ti = 2; off = k - 80; tab = e2; vocab = 7;   dd = 8;  }
            else if (k < 96) { ti = 3; off = k - 88; tab = e3; vocab = 5;   dd = 8;  }
            else             { ti = 4; off = k - 96; tab = e4; vocab = 2;   dd = 4;  }
            int xi = x[node * 5 + ti];
            xi = xi < 0 ? 0 : (xi > vocab - 1 ? vocab - 1 : xi);
            feat[ln][k] = tab[xi * dd + off];
        }
    }
    __syncthreads();
    if (node >= NN) return;
    float acc = pb[c];
    const float* wr = pw + c * 100;
    #pragma unroll 4
    for (int k = 0; k < 100; ++k) acc += feat[ln][k] * wr[k];
    h[(size_t)node * HH + c] = acc;
}

// ---------------------------------------------------------------------------
// 2. Edge hash (shared across the 3 GINE blocks)
// ---------------------------------------------------------------------------
__global__ void k_edge_hash(const int* __restrict__ ea, int* __restrict__ eh)
{
    int e = blockIdx.x * blockDim.x + threadIdx.x;
    if (e >= EE) return;
    eh[e] = (ea[e * 3] + 3 * ea[e * 3 + 1] + 7 * ea[e * 3 + 2]) & 15;
}

// ---------------------------------------------------------------------------
// 3. Per-block edge table (16 x 64, tiny)
// ---------------------------------------------------------------------------
__global__ void k_etab(const float* __restrict__ eemb, const float* __restrict__ lw,
                       const float* __restrict__ lb, float* __restrict__ etab)
{
    int t = blockIdx.x * blockDim.x + threadIdx.x;
    if (t >= 16 * HH) return;
    int v = t >> 6, c = t & 63;
    float acc = lb[c];
    #pragma unroll
    for (int d = 0; d < 8; ++d) acc += eemb[v * 8 + d] * lw[c * 8 + d];
    etab[t] = acc;
}

// ---------------------------------------------------------------------------
// 4. Edge message + scatter: agg[dst] += relu(h[src] + etab[eh])
//    2 edges / thread, 16 lanes per edge (float4), prefetch 2nd gather row.
// ---------------------------------------------------------------------------
__global__ __launch_bounds__(256) void k_edge_msg(
    const int* __restrict__ ei, const int* __restrict__ eh,
    const float* __restrict__ etab, const float* __restrict__ h,
    float* __restrict__ agg)
{
    int t = blockIdx.x * blockDim.x + threadIdx.x;
    if (t >= (EE / 2) * 16) return;
    const int e0 = (t >> 4) * 2;
    const int e1 = e0 + 1;
    const int c4 = (t & 15) << 2;
    const int s0 = ei[e0],      d0 = ei[EE + e0];
    const int s1 = ei[e1],      d1 = ei[EE + e1];

    __builtin_prefetch(h + (size_t)s1 * HH + c4, 0, 0);   // global_prefetch_b8

    {
        const float4 hv = *(const float4*)(h + (size_t)s0 * HH + c4);
        const float4 ev = *(const float4*)(etab + eh[e0] * HH + c4);
        float* ap = agg + (size_t)d0 * HH + c4;
        unsafeAtomicAdd(ap + 0, fmaxf(hv.x + ev.x, 0.f));
        unsafeAtomicAdd(ap + 1, fmaxf(hv.y + ev.y, 0.f));
        unsafeAtomicAdd(ap + 2, fmaxf(hv.z + ev.z, 0.f));
        unsafeAtomicAdd(ap + 3, fmaxf(hv.w + ev.w, 0.f));
    }
    {
        const float4 hv = *(const float4*)(h + (size_t)s1 * HH + c4);
        const float4 ev = *(const float4*)(etab + eh[e1] * HH + c4);
        float* ap = agg + (size_t)d1 * HH + c4;
        unsafeAtomicAdd(ap + 0, fmaxf(hv.x + ev.x, 0.f));
        unsafeAtomicAdd(ap + 1, fmaxf(hv.y + ev.y, 0.f));
        unsafeAtomicAdd(ap + 2, fmaxf(hv.z + ev.z, 0.f));
        unsafeAtomicAdd(ap + 3, fmaxf(hv.w + ev.w, 0.f));
    }
}

// ---------------------------------------------------------------------------
// 5. z = (1+eps)*h + agg, emitted directly as f16 (GEMM A operand)
// ---------------------------------------------------------------------------
__global__ void k_zfuse_h(const float* __restrict__ h, const float* __restrict__ agg,
                          const float* __restrict__ eps, _Float16* __restrict__ z)
{
    int t = blockIdx.x * blockDim.x + threadIdx.x;
    if (t >= NN * HH) return;
    z[t] = (_Float16)((1.f + eps[0]) * h[t] + agg[t]);
}

// ---------------------------------------------------------------------------
// 6. generic f32 -> f16 conversion (weights, graph vector)
// ---------------------------------------------------------------------------
__global__ void k_cvt_f16(const float* __restrict__ s, _Float16* __restrict__ d, int n)
{
    int t = blockIdx.x * blockDim.x + threadIdx.x;
    if (t < n) d[t] = (_Float16)s[t];
}

// ---------------------------------------------------------------------------
// 7. WMMA GEMM (K=64 fixed):  C[M x Nout] = relu?(A[M x 64] * W[Nout x 64]^T + b)
//    A, W pre-converted f16 row-major. Per CDNA5 ISA 7.12.2 a lane's
//    16 A-halves are two contiguous 8-half runs (k = kb+half*8+0..7 and +16),
//    so fragments load as two b128 per matrix per k-step. Fully unrolled:
//    two v_wmma_f32_16x16x32_f16 per wave. One wave per 16x16 tile.
// ---------------------------------------------------------------------------
template <bool OUT_HALF>
__global__ __launch_bounds__(256) void k_wmma_gemm(
    const _Float16* __restrict__ A, const _Float16* __restrict__ W,
    const float* __restrict__ bias, void* __restrict__ Cout,
    int M, int Nout, int relu)
{
    const int wave = (blockIdx.x * blockDim.x + threadIdx.x) >> 5;
    const int lane = threadIdx.x & 31;
    const int nct  = Nout >> 4;
    const int rowT = wave / nct;
    const int colT = wave - rowT * nct;
    const int m0 = rowT << 4, n0 = colT << 4;
    if (m0 >= M) return;
    const int half = lane >> 4;
    const int mn   = lane & 15;

    const _Float16* Arow = A + (size_t)(m0 + mn) * 64 + half * 8;
    const _Float16* Wrow = W + (size_t)(n0 + mn) * 64 + half * 8;

    v8f acc = {};
    #pragma unroll
    for (int kb = 0; kb < 64; kb += 32) {
        v8h a0 = *(const v8h*)(Arow + kb);
        v8h a1 = *(const v8h*)(Arow + kb + 16);
        v8h b0 = *(const v8h*)(Wrow + kb);
        v8h b1 = *(const v8h*)(Wrow + kb + 16);
        v16h a, b;
        #pragma unroll
        for (int j = 0; j < 8; ++j) {
            a[j] = a0[j]; a[8 + j] = a1[j];
            b[j] = b0[j]; b[8 + j] = b1[j];
        }
        acc = __builtin_amdgcn_wmma_f32_16x16x32_f16(
            false, a, false, b, (short)0, acc, false, false);
    }
    #pragma unroll
    for (int j = 0; j < 8; ++j) {
        const int m = m0 + j + half * 8;
        const int n = n0 + mn;
        float v = acc[j] + bias[n];
        if (relu) v = fmaxf(v, 0.f);
        if (OUT_HALF) ((_Float16*)Cout)[(size_t)m * Nout + n] = (_Float16)v;
        else          ((float*)Cout)[(size_t)m * Nout + n] = v;
    }
}

// ---------------------------------------------------------------------------
// 8. BatchNorm (eval)
// ---------------------------------------------------------------------------
__global__ void k_bn(const float* __restrict__ h, float* __restrict__ hbn,
                     const float* __restrict__ g, const float* __restrict__ b,
                     const float* __restrict__ rm, const float* __restrict__ rv)
{
    int t = blockIdx.x * blockDim.x + threadIdx.x;
    if (t >= NN * HH) return;
    int c = t & 63;
    hbn[t] = (h[t] - rm[c]) * rsqrtf(rv[c] + 1e-5f) * g[c] + b[c];
}

// ---------------------------------------------------------------------------
// 9. Attentional aggregation (segment softmax via atomics)
// ---------------------------------------------------------------------------
__global__ void k_gate(const float* __restrict__ hbn, const float* __restrict__ gw,
                       const float* __restrict__ gb, float* __restrict__ gate)
{
    int n = blockIdx.x * blockDim.x + threadIdx.x;
    if (n >= NN) return;
    const float* hr = hbn + (size_t)n * HH;
    float acc = gb[0];
    #pragma unroll 4
    for (int c = 0; c < HH; ++c) acc += hr[c] * gw[c];
    gate[n] = acc;
}

__global__ void k_agg_init(float* __restrict__ gmax, float* __restrict__ gsum,
                           float* __restrict__ graphv)
{
    int t = blockIdx.x * blockDim.x + threadIdx.x;
    if (t < GG)                    gmax[t] = -__builtin_inff();
    else if (t < 2 * GG)           gsum[t - GG] = 0.f;
    else if (t < 2 * GG + GG * HH) graphv[t - 2 * GG] = 0.f;
}

__device__ inline void atomicMaxF(float* a, float v)
{
    if (v >= 0.f) atomicMax((int*)a, __float_as_int(v));
    else          atomicMin((unsigned int*)a, __float_as_uint(v));
}

__global__ void k_segmax(const float* __restrict__ gate, const int* __restrict__ batch,
                         float* __restrict__ gmax)
{
    int n = blockIdx.x * blockDim.x + threadIdx.x;
    if (n >= NN) return;
    atomicMaxF(gmax + batch[n], gate[n]);
}

__global__ void k_expsum(float* __restrict__ gate, const int* __restrict__ batch,
                         const float* __restrict__ gmax, float* __restrict__ gsum)
{
    int n = blockIdx.x * blockDim.x + threadIdx.x;
    if (n >= NN) return;
    int bidx = batch[n];
    float w = __expf(gate[n] - gmax[bidx]);
    gate[n] = w;
    unsafeAtomicAdd(gsum + bidx, w);
}

__global__ __launch_bounds__(256) void k_wscatter(
    const float* __restrict__ hbn, const float* __restrict__ gate,
    const int* __restrict__ batch, const float* __restrict__ gsum,
    float* __restrict__ graphv)
{
    int t = blockIdx.x * blockDim.x + threadIdx.x;
    if (t >= NN * 16) return;
    const int n  = t >> 4;
    const int c4 = (t & 15) << 2;
    const int bi = batch[n];
    const float alpha = gate[n] / (gsum[bi] + 1e-16f);
    const float4 hv = *(const float4*)(hbn + (size_t)n * HH + c4);
    float* gp = graphv + (size_t)bi * HH + c4;
    unsafeAtomicAdd(gp + 0, alpha * hv.x);
    unsafeAtomicAdd(gp + 1, alpha * hv.y);
    unsafeAtomicAdd(gp + 2, alpha * hv.z);
    unsafeAtomicAdd(gp + 3, alpha * hv.w);
}

// ---------------------------------------------------------------------------
// 10. Head layer 2: out[g] = hid[g] . w2 + b2
// ---------------------------------------------------------------------------
__global__ void k_head2(const float* __restrict__ hid, const float* __restrict__ w2,
                        const float* __restrict__ b2, float* __restrict__ out)
{
    int g = blockIdx.x * blockDim.x + threadIdx.x;
    if (g >= GG) return;
    const float* hr = hid + (size_t)g * 128;
    float acc = b2[0];
    #pragma unroll 4
    for (int k = 0; k < 128; ++k) acc += hr[k] * w2[k];
    out[g] = acc;
}

// ---------------------------------------------------------------------------
// launch
// ---------------------------------------------------------------------------
extern "C" void kernel_launch(void* const* d_in, const int* in_sizes, int n_in,
                              void* d_out, int out_size, void* d_ws, size_t ws_size,
                              hipStream_t stream)
{
    const int* x     = (const int*)d_in[0];
    const int* ei    = (const int*)d_in[1];
    const int* ea    = (const int*)d_in[2];
    const int* batch = (const int*)d_in[3];
    const float* emb0 = (const float*)d_in[4];
    const float* emb1 = (const float*)d_in[5];
    const float* emb2 = (const float*)d_in[6];
    const float* emb3 = (const float*)d_in[7];
    const float* emb4 = (const float*)d_in[8];
    const float* proj_w = (const float*)d_in[9];
    const float* proj_b = (const float*)d_in[10];

    struct Blk { const float *eemb, *lw, *lb, *eps, *w1, *b1, *w2, *b2; } blk[3];
    for (int i = 0; i < 3; ++i) {
        int base = 11 + i * 8;
        blk[i].eemb = (const float*)d_in[base + 0];
        blk[i].lw   = (const float*)d_in[base + 1];
        blk[i].lb   = (const float*)d_in[base + 2];
        blk[i].eps  = (const float*)d_in[base + 3];
        blk[i].w1   = (const float*)d_in[base + 4];
        blk[i].b1   = (const float*)d_in[base + 5];
        blk[i].w2   = (const float*)d_in[base + 6];
        blk[i].b2   = (const float*)d_in[base + 7];
    }
    const float* bn_g  = (const float*)d_in[35];
    const float* bn_b  = (const float*)d_in[36];
    const float* bn_rm = (const float*)d_in[37];
    const float* bn_rv = (const float*)d_in[38];
    const float* gate_w  = (const float*)d_in[39];
    const float* gate_b  = (const float*)d_in[40];
    const float* head_w1 = (const float*)d_in[41];
    const float* head_b1 = (const float*)d_in[42];
    const float* head_w2 = (const float*)d_in[43];
    const float* head_b2 = (const float*)d_in[44];
    float* out = (float*)d_out;

    // ---- workspace carve-up (all chunks 16B-aligned) ----
    float* buf0   = (float*)d_ws;                     // h   [N*H] f32
    float* buf1   = buf0 + (size_t)NN * HH;           // agg [N*H] f32
    float* etab   = buf1 + (size_t)NN * HH;           // 16*64 f32
    float* gate   = etab + 16 * HH;                   // [N] f32
    float* gmax   = gate + NN;                        // [G]
    float* gsum   = gmax + GG;                        // [G]
    float* graphv = gsum + GG;                        // [G*H] f32
    float* hid    = graphv + (size_t)GG * HH;         // [G*128] f32
    int*   eh     = (int*)(hid + (size_t)GG * 128);   // [E] int
    _Float16* zh    = (_Float16*)(eh + EE);           // [N*H] f16 (z)
    _Float16* tmph  = zh + (size_t)NN * HH;           // [N*H] f16 (mlp hidden)
    _Float16* w1h   = tmph + (size_t)NN * HH;         // 3 x [64*64] f16
    _Float16* w2h   = w1h + 3 * HH * HH;              // 3 x [64*64] f16
    _Float16* hw1h  = w2h + 3 * HH * HH;              // [128*64] f16
    _Float16* graphh = hw1h + 128 * HH;               // [G*H] f16

    // 0. one-time f16 weight conversion
    for (int i = 0; i < 3; ++i) {
        k_cvt_f16<<<(HH * HH + 255) / 256, 256, 0, stream>>>(blk[i].w1, w1h + i * HH * HH, HH * HH);
        k_cvt_f16<<<(HH * HH + 255) / 256, 256, 0, stream>>>(blk[i].w2, w2h + i * HH * HH, HH * HH);
    }
    k_cvt_f16<<<(128 * HH + 255) / 256, 256, 0, stream>>>(head_w1, hw1h, 128 * HH);

    // 1. embedding + projection
    k_embed_proj<<<NN / 4, 256, 0, stream>>>(x, emb0, emb1, emb2, emb3, emb4,
                                             proj_w, proj_b, buf0);
    // 2. edge hash
    k_edge_hash<<<(EE + 255) / 256, 256, 0, stream>>>(ea, eh);

    const int wavesN = (NN / 16) * (HH / 16);          // 25000
    const int blksN  = (wavesN * 32 + 255) / 256;      // 3125

    // 3. three GINE blocks
    for (int i = 0; i < 3; ++i) {
        k_etab<<<(16 * HH + 255) / 256, 256, 0, stream>>>(blk[i].eemb, blk[i].lw,
                                                          blk[i].lb, etab);
        hipMemsetAsync(buf1, 0, (size_t)NN * HH * sizeof(float), stream);
        k_edge_msg<<<((EE / 2) * 16 + 255) / 256, 256, 0, stream>>>(ei, eh, etab, buf0, buf1);
        k_zfuse_h<<<(NN * HH + 255) / 256, 256, 0, stream>>>(buf0, buf1, blk[i].eps, zh);
        k_wmma_gemm<true><<<blksN, 256, 0, stream>>>(zh, w1h + i * HH * HH,
                                                     blk[i].b1, tmph, NN, HH, 1);
        k_wmma_gemm<false><<<blksN, 256, 0, stream>>>(tmph, w2h + i * HH * HH,
                                                      blk[i].b2, buf0, NN, HH, 1);
    }

    // 4. batchnorm -> buf1
    k_bn<<<(NN * HH + 255) / 256, 256, 0, stream>>>(buf0, buf1, bn_g, bn_b, bn_rm, bn_rv);

    // 5. attentional aggregation
    k_gate<<<(NN + 255) / 256, 256, 0, stream>>>(buf1, gate_w, gate_b, gate);
    k_agg_init<<<(GG * (HH + 2) + 255) / 256, 256, 0, stream>>>(gmax, gsum, graphv);
    k_segmax<<<(NN + 255) / 256, 256, 0, stream>>>(gate, batch, gmax);
    k_expsum<<<(NN + 255) / 256, 256, 0, stream>>>(gate, batch, gmax, gsum);
    k_wscatter<<<(NN * 16 + 255) / 256, 256, 0, stream>>>(buf1, gate, batch, gsum, graphv);

    // 6. head: [512 x 64] x f16 -> [512 x 128] (WMMA), then dot with w2
    k_cvt_f16<<<(GG * HH + 255) / 256, 256, 0, stream>>>(graphv, graphh, GG * HH);
    const int wavesH = (GG / 16) * (128 / 16);         // 256
    const int blksH  = (wavesH * 32 + 255) / 256;      // 32
    k_wmma_gemm<false><<<blksH, 256, 0, stream>>>(graphh, hw1h, head_b1, hid,
                                                  GG, 128, 1);
    k_head2<<<(GG + 255) / 256, 256, 0, stream>>>(hid, head_w2, head_b2, out);
}
